// LSTMInitialisationPolicy_4054449127520
// MI455X (gfx1250) — matrix operand order repaired and checked by reading, not compiled, not run on hardware
//
#include <hip/hip_runtime.h>
#include <hip/hip_bf16.h>
#include <stdint.h>

// ============================================================================
// Sequential 2-layer LSTM sampler for MI455X (gfx1250).
//  - Weights converted once to bf16 (24 MB, L2-resident on a 192MB-L2 part).
//  - 64 persistent waves (16 blocks x 4 waves); wave a owns hidden slice
//    j in [16a,16a+16) and its 4 gate-row tiles in every matrix.
//  - v_wmma_f32_16x16x32_bf16 mat-vec with h replicated across the 16 B
//    columns (memory-bound step, so the N-waste is free); every lane then
//    holds valid gate values -> cell state c0/c1 lives in registers.
//  - TDM (tensor_load_to_lds, 6-arg form on this toolchain) broadcasts the
//    h vectors into LDS each phase; waits via s_wait_tensorcnt.
//  - 2 atomic grid barriers per step; h0/h1 double-buffered bf16 in d_ws.
//  - Needs ~24.2 MB of workspace.
// ============================================================================

typedef unsigned short u16;
typedef unsigned int   u32;
typedef __attribute__((ext_vector_type(16))) __bf16 v16bf;
typedef __attribute__((ext_vector_type(8)))  float  v8f;
typedef __attribute__((ext_vector_type(4)))  u32    v4u;
typedef __attribute__((ext_vector_type(8)))  int    v8i;
typedef __attribute__((ext_vector_type(4)))  int    v4i;

#define HDIM   1024
#define TSTEPS 8192
#define CTX    512
#define NBLK   16
#define TPB    128     // 4 waves/block -> 64 waves total = 64 j-tiles of 16

// ---- workspace layout (bytes) ----
#define OFF_WHH0  ((size_t)0)                 // bf16 4096x1024 (8MB)
#define OFF_WIH1  ((size_t)8u  << 20)         // bf16 4096x1024
#define OFF_WHH1  ((size_t)16u << 20)         // bf16 4096x1024
#define OFF_W2    ((size_t)24u << 20)         // bf16 1024 (2KB)
#define OFF_B0S   (OFF_W2   + 4096)           // f32 4096 = bih0+bhh0
#define OFF_B1S   (OFF_B0S  + 16384)          // f32 4096 = bih1+bhh1
#define OFF_H0BUF (OFF_B1S  + 16384)          // bf16 2x1024 (double buffer)
#define OFF_H1BUF (OFF_H0BUF + 4096)          // bf16 2x1024
#define OFF_BAR   (OFF_H1BUF + 4096)          // u32[2] grid barrier

__device__ __forceinline__ u16 f2bf(float f) {            // RNE f32->bf16
  u32 x = __float_as_uint(f);
  x += 0x7FFFu + ((x >> 16) & 1u);
  return (u16)(x >> 16);
}
__device__ __forceinline__ float bf2f(u16 b) { return __uint_as_float(((u32)b) << 16); }
__device__ __forceinline__ float sigmoidf_(float z) { return 1.0f / (1.0f + __expf(-z)); }
__device__ __forceinline__ u32 rotl_(u32 x, int r) { return (x << r) | (x >> (32 - r)); }

// Threefry2x32-20 (jax.random PRNG core)
__device__ __forceinline__ void threefry2x32(u32 k0, u32 k1, u32& x0, u32& x1) {
  u32 ks[3] = { k0, k1, 0x1BD11BDAu ^ k0 ^ k1 };
  const int RA[4] = {13, 15, 26, 6}, RB[4] = {17, 29, 16, 24};
  x0 += ks[0]; x1 += ks[1];
#pragma unroll
  for (int i = 0; i < 5; ++i) {
    const int* R = (i & 1) ? RB : RA;
#pragma unroll
    for (int j = 0; j < 4; ++j) { x0 += x1; x1 = rotl_(x1, R[j]); x1 ^= x0; }
    x0 += ks[(i + 1) % 3];
    x1 += ks[(i + 2) % 3] + (u32)(i + 1);
  }
}

// ---- WMMA fragment helpers (bf16 16x16x32, layouts per ISA 7.12.2) ----
union FragU { v16bf v; v4u q[2]; };
__device__ __forceinline__ v16bf ldfrag(const u16* p0, const u16* p1) {
  FragU f;
  __builtin_memcpy(&f.q[0], p0, 16);
  __builtin_memcpy(&f.q[1], p1, 16);
  return f.v;
}
#define WMMA_BF16(A, B, C) \
  __builtin_amdgcn_wmma_f32_16x16x32_bf16(false, (A), false, (B), (short)0, (C), false, false)

// ---- TDM: 1-row tensor load of `nelem` bf16 elements into LDS ----
// 6-arg builtin on this toolchain: (v4u g0, v8i g1, v4i g2, v4i g3, v8i, i32 cpol)
__device__ __forceinline__ void tdm_load_row(u32 lds_off, const void* gptr, u32 nelem) {
  unsigned long long ga = (unsigned long long)(uintptr_t)gptr;
  v4u g0;
  g0.x = 1u;                                        // count = 1 descriptor
  g0.y = lds_off;                                   // lds_addr
  g0.z = (u32)ga;                                   // global_addr[31:0]
  g0.w = (u32)((ga >> 32) & 0x01FFFFFFu) | 0x80000000u; // addr[56:32] | type=2
  v8i g1;
  g1[0] = (int)(1u << 16);                          // data_size=1 (2B), no mask
  g1[1] = (int)(nelem << 16);                       // tensor_dim0 lo
  g1[2] = (int)((nelem >> 16) | (1u << 16));        // tensor_dim0 hi | tensor_dim1=1
  g1[3] = (int)(nelem << 16);                       // tile_dim0
  g1[4] = 1;                                        // tile_dim1 = 1
  g1[5] = (int)nelem;                               // tensor_dim0_stride lo
  g1[6] = 0; g1[7] = 0;
  v4i z4 = {0, 0, 0, 0};
  v8i z8 = {0, 0, 0, 0, 0, 0, 0, 0};
  __builtin_amdgcn_tensor_load_to_lds(g0, g1, z4, z4, z8, 0);
}

// ---- grid-wide sense barrier (agent scope) ----
__device__ __forceinline__ void grid_barrier(u32* bar, u32& epoch) {
  __threadfence();                // every wave flushes its own stores
  __syncthreads();
  if (threadIdx.x == 0) {
    u32 prev = __hip_atomic_fetch_add(&bar[0], 1u, __ATOMIC_ACQ_REL, __HIP_MEMORY_SCOPE_AGENT);
    if (prev == (u32)gridDim.x - 1u) {
      __hip_atomic_store(&bar[0], 0u, __ATOMIC_RELAXED, __HIP_MEMORY_SCOPE_AGENT);
      __hip_atomic_fetch_add(&bar[1], 1u, __ATOMIC_RELEASE, __HIP_MEMORY_SCOPE_AGENT);
    } else {
      while (__hip_atomic_load(&bar[1], __ATOMIC_ACQUIRE, __HIP_MEMORY_SCOPE_AGENT) <= epoch)
        __builtin_amdgcn_s_sleep(2);
    }
  }
  epoch++;
  __syncthreads();
}

// ============================================================================
// prep: fp32 -> bf16 weights, bias sums, zero h buffers + barrier
// ============================================================================
__global__ void prep_kernel(const float* __restrict__ Whh0, const float* __restrict__ Wih1,
                            const float* __restrict__ Whh1, const float* __restrict__ W2,
                            const float* __restrict__ bih0, const float* __restrict__ bhh0,
                            const float* __restrict__ bih1, const float* __restrict__ bhh1,
                            unsigned char* __restrict__ ws) {
  u16* w0  = (u16*)(ws + OFF_WHH0);
  u16* w1  = (u16*)(ws + OFF_WIH1);
  u16* w2  = (u16*)(ws + OFF_WHH1);
  u16* w2v = (u16*)(ws + OFF_W2);
  float* b0s = (float*)(ws + OFF_B0S);
  float* b1s = (float*)(ws + OFF_B1S);
  u16* h0  = (u16*)(ws + OFF_H0BUF);
  u16* h1  = (u16*)(ws + OFF_H1BUF);
  u32* bar = (u32*)(ws + OFF_BAR);
  size_t i0 = (size_t)blockIdx.x * blockDim.x + threadIdx.x;
  size_t st = (size_t)gridDim.x * blockDim.x;
  for (size_t i = i0; i < (size_t)4096 * 1024; i += st) {
    w0[i] = f2bf(Whh0[i]); w1[i] = f2bf(Wih1[i]); w2[i] = f2bf(Whh1[i]);
  }
  for (size_t i = i0; i < 1024; i += st) w2v[i] = f2bf(W2[i]);
  for (size_t i = i0; i < 4096; i += st) { b0s[i] = bih0[i] + bhh0[i]; b1s[i] = bih1[i] + bhh1[i]; }
  for (size_t i = i0; i < 2048; i += st) { h0[i] = 0; h1[i] = 0; }
  if (i0 < 2) bar[i0] = 0;
}

// ============================================================================
// persistent sequential LSTM kernel
// ============================================================================
__global__ void __launch_bounds__(TPB, 1)
lstm_kernel(const float* __restrict__ context, const float* __restrict__ W1,
            const float* __restrict__ b1p, const float* __restrict__ Wih0,
            const float* __restrict__ b2p, unsigned char* __restrict__ ws,
            float* __restrict__ out) {
  __shared__ __align__(16) u16 lds_h0[HDIM];
  __shared__ __align__(16) u16 lds_h1[HDIM];

  const u16* Whh0b = (const u16*)(ws + OFF_WHH0);
  const u16* Wih1b = (const u16*)(ws + OFF_WIH1);
  const u16* Whh1b = (const u16*)(ws + OFF_WHH1);
  const u16* W2b   = (const u16*)(ws + OFF_W2);
  const float* b0s = (const float*)(ws + OFF_B0S);
  const float* b1s = (const float*)(ws + OFF_B1S);
  u16* h0buf = (u16*)(ws + OFF_H0BUF);
  u16* h1buf = (u16*)(ws + OFF_H1BUF);
  u32* bar   = (u32*)(ws + OFF_BAR);

  const int tid  = threadIdx.x;
  const int lane = tid & 31;
  const int warp = tid >> 5;
  const int a    = blockIdx.x * (TPB / 32) + warp;   // j-tile id 0..63
  const int rsel = (lane >> 4) & 1;                  // which K half-group
  const int rrow = lane & 15;                        // A-matrix row in tile
  const int jb   = a * 16 + rsel * 8;                // first hidden index owned via VGPR r

  const u32 lds0off = (u32)(uintptr_t)(&lds_h0[0]);
  const u32 lds1off = (u32)(uintptr_t)(&lds_h1[0]);

  // per-lane row offsets (elements) for the 4 gate tiles in each 4096x1024 matrix
  size_t rowoff[4];
#pragma unroll
  for (int g = 0; g < 4; ++g) rowoff[g] = (size_t)(g * HDIM + a * 16 + rrow) * HDIM;

  // x0 = W1 @ context + b1  (every lane/wave computes identical value)
  float part = 0.f;
#pragma unroll 4
  for (int i = 0; i < CTX / 32; ++i) { int k = lane * (CTX / 32) + i; part += W1[k] * context[k]; }
#pragma unroll
  for (int m = 16; m > 0; m >>= 1) part += __shfl_xor(part, m, 32);
  float x = part + b1p[0];
  const float b2v = b2p[0];

  float c0[8], c1[8];
#pragma unroll
  for (int r = 0; r < 8; ++r) { c0[r] = 0.f; c1[r] = 0.f; }
  float logp = 0.f;
  u32 epoch = 0;
  int par = 0;
  const bool writer = (blockIdx.x == 0 && tid == 0);

  for (int t = 0; t < TSTEPS; ++t) {
    // ---------- phase A: stage h0,h1 into LDS via TDM ----------
    if (warp == 0) {
      tdm_load_row(lds0off, h0buf + (size_t)par * HDIM, HDIM);
      tdm_load_row(lds1off, h1buf + (size_t)par * HDIM, HDIM);
      __builtin_amdgcn_s_wait_tensorcnt(0);
    }
    __syncthreads();

    v8f acc0[4], accu[4];
#pragma unroll
    for (int g = 0; g < 4; ++g) { acc0[g] = (v8f)0.0f; accu[g] = (v8f)0.0f; }

#pragma unroll 2
    for (int kb = 0; kb < HDIM; kb += 32) {
      const u16* hb0 = lds_h0 + kb + rsel * 8;
      const u16* hb1 = lds_h1 + kb + rsel * 8;
      v16bf bf0 = ldfrag(hb0, hb0 + 16);   // h replicated over all 16 N columns
      v16bf bf1 = ldfrag(hb1, hb1 + 16);
#pragma unroll
      for (int g = 0; g < 4; ++g) {
        const u16* p = Whh0b + rowoff[g] + kb + rsel * 8;
        acc0[g] = WMMA_BF16(ldfrag(p, p + 16), bf0, acc0[g]);   // Whh0 @ h0
        const u16* q = Whh1b + rowoff[g] + kb + rsel * 8;
        accu[g] = WMMA_BF16(ldfrag(q, q + 16), bf1, accu[g]);   // Whh1 @ h1
      }
    }

    // cell 0 (every lane holds a valid copy; j = jb + r)
    float hn[8];
#pragma unroll
    for (int r = 0; r < 8; ++r) {
      int j = jb + r;
      float pi = acc0[0][r] + b0s[j]            + Wih0[j]            * x;
      float pf = acc0[1][r] + b0s[HDIM + j]     + Wih0[HDIM + j]     * x;
      float pg = acc0[2][r] + b0s[2 * HDIM + j] + Wih0[2 * HDIM + j] * x;
      float po = acc0[3][r] + b0s[3 * HDIM + j] + Wih0[3 * HDIM + j] * x;
      float iv = sigmoidf_(pi), fv = sigmoidf_(pf), gv = tanhf(pg), ov = sigmoidf_(po);
      c0[r] = fv * c0[r] + iv * gv;
      hn[r] = ov * tanhf(c0[r]);
    }
    if (rrow == 0) {            // lanes 0 and 16 publish the 16 new h0 values
      u16 pk[8];
#pragma unroll
      for (int r = 0; r < 8; ++r) pk[r] = f2bf(hn[r]);
      __builtin_memcpy(h0buf + (size_t)(par ^ 1) * HDIM + jb, pk, 16);
    }
    grid_barrier(bar, epoch);

    // ---------- phase B: Wih1 @ h0_new ----------
    if (warp == 0) {
      tdm_load_row(lds0off, h0buf + (size_t)(par ^ 1) * HDIM, HDIM);
      __builtin_amdgcn_s_wait_tensorcnt(0);
    }
    __syncthreads();

    v8f acc1[4];
#pragma unroll
    for (int g = 0; g < 4; ++g) acc1[g] = (v8f)0.0f;
#pragma unroll 2
    for (int kb = 0; kb < HDIM; kb += 32) {
      const u16* hb = lds_h0 + kb + rsel * 8;
      v16bf bfh = ldfrag(hb, hb + 16);
#pragma unroll
      for (int g = 0; g < 4; ++g) {
        const u16* p = Wih1b + rowoff[g] + kb + rsel * 8;
        acc1[g] = WMMA_BF16(ldfrag(p, p + 16), bfh, acc1[g]);
      }
    }

    // cell 1: gates = Wih1@h0_new + Whh1@h1(prev, in accu) + b1
#pragma unroll
    for (int r = 0; r < 8; ++r) {
      int j = jb + r;
      float pi = acc1[0][r] + accu[0][r] + b1s[j];
      float pf = acc1[1][r] + accu[1][r] + b1s[HDIM + j];
      float pg = acc1[2][r] + accu[2][r] + b1s[2 * HDIM + j];
      float po = acc1[3][r] + accu[3][r] + b1s[3 * HDIM + j];
      float iv = sigmoidf_(pi), fv = sigmoidf_(pf), gv = tanhf(pg), ov = sigmoidf_(po);
      c1[r] = fv * c1[r] + iv * gv;
      hn[r] = ov * tanhf(c1[r]);
    }
    if (rrow == 0) {
      u16 pk[8];
#pragma unroll
      for (int r = 0; r < 8; ++r) pk[r] = f2bf(hn[r]);
      __builtin_memcpy(h1buf + (size_t)(par ^ 1) * HDIM + jb, pk, 16);
    }
    grid_barrier(bar, epoch);

    // ---------- phase C: p = sigmoid(W2@h1+b2); Bernoulli; logp ----------
    // Redundantly computed bit-identically by every wave -> no broadcast sync.
    const u16* h1n = h1buf + (size_t)(par ^ 1) * HDIM;
    float d = 0.f;
#pragma unroll 8
    for (int i = 0; i < 32; ++i) { int k = lane * 32 + i; d += bf2f(W2b[k]) * bf2f(h1n[k]); }
#pragma unroll
    for (int m = 16; m > 0; m >>= 1) d += __shfl_xor(d, m, 32);
    float p = sigmoidf_(d + b2v);

    u32 k0 = (u32)(2 * t), k1 = (u32)(2 * t + 1);
    threefry2x32(0u, 42u, k0, k1);           // derive per-step key from key(42)
    u32 r0 = 0u, r1 = 0u;
    threefry2x32(k0, k1, r0, r1);            // random bits
    float u = __uint_as_float(0x3f800000u | (r0 >> 9)) - 1.0f;
    float s = (u < p) ? 1.0f : 0.0f;
    logp += s * __logf(p) + (1.0f - s) * log1pf(-p);
    if (writer) out[t] = s;
    x = s;
    par ^= 1;
  }
  if (writer) out[TSTEPS] = logp;
}

// ============================================================================
extern "C" void kernel_launch(void* const* d_in, const int* in_sizes, int n_in,
                              void* d_out, int out_size, void* d_ws, size_t ws_size,
                              hipStream_t stream) {
  const float* context = (const float*)d_in[0];
  const float* W1      = (const float*)d_in[1];
  const float* b1      = (const float*)d_in[2];
  const float* Wih0    = (const float*)d_in[3];
  const float* Whh0    = (const float*)d_in[4];
  const float* bih0    = (const float*)d_in[5];
  const float* bhh0    = (const float*)d_in[6];
  const float* Wih1    = (const float*)d_in[7];
  const float* Whh1    = (const float*)d_in[8];
  const float* bih1    = (const float*)d_in[9];
  const float* bhh1    = (const float*)d_in[10];
  const float* W2      = (const float*)d_in[11];
  const float* b2      = (const float*)d_in[12];
  float* out = (float*)d_out;
  unsigned char* ws = (unsigned char*)d_ws;

  prep_kernel<<<1024, 256, 0, stream>>>(Whh0, Wih1, Whh1, W2, bih0, bhh0, bih1, bhh1, ws);
  lstm_kernel<<<NBLK, TPB, 0, stream>>>(context, W1, b1, Wih0, b2, ws, out);
}
